// Model_73667279061581
// MI455X (gfx1250) — matrix operand order repaired and checked by reading, not compiled, not run on hardware
//
#include <hip/hip_runtime.h>
#include <hip/hip_bf16.h>

typedef __attribute__((ext_vector_type(16))) __bf16    v16bf;
typedef __attribute__((ext_vector_type(8)))  float     v8f;
typedef __attribute__((ext_vector_type(8)))  unsigned  v8u;

// ---------- helpers ----------

__device__ __forceinline__ unsigned short f32_bf16(float f) {
    unsigned u = __builtin_bit_cast(unsigned, f);
    u += 0x7FFFu + ((u >> 16) & 1u);           // round-to-nearest-even
    return (unsigned short)(u >> 16);
}

__device__ __forceinline__ unsigned pack_bf16(float a, float b) {
    return (unsigned)f32_bf16(a) | ((unsigned)f32_bf16(b) << 16);
}

// 16B global -> LDS async copy (CDNA5 async path, ASYNCcnt-tracked)
__device__ __forceinline__ void async_copy16(unsigned lds_off, const void* gaddr) {
    asm volatile("global_load_async_to_lds_b128 %0, %1, off"
                 :: "v"(lds_off), "v"(gaddr) : "memory");
}
__device__ __forceinline__ void wait_async0() {
    asm volatile("s_wait_asynccnt 0x0" ::: "memory");
}

// stage contiguous bf16 block global -> LDS with async 16B copies
__device__ __forceinline__ void async_stage(const unsigned short* __restrict__ src,
                                            unsigned short* dst, int nush, int tid) {
    const unsigned base = (unsigned)(uintptr_t)dst;
    for (int i = tid * 8; i < nush; i += 256 * 8)
        async_copy16(base + 2u * (unsigned)i, src + i);
}

__device__ __forceinline__ v8f wmma_bf16(v16bf a, v16bf b, v8f c) {
    return __builtin_amdgcn_wmma_f32_16x16x32_bf16(
        false, a, false, b, (short)0, c, false, false);
}

// A fragment: 16x32 bf16, lane = M (mod 16):
//   k = kbase + (r&3)*2 + (r>>2)*16 + (lane>=16 ? 8 : 0)
__device__ __forceinline__ v16bf load_A(const unsigned short* X, int stride,
                                        int m, int kbase, int lane) {
    const int half8 = ((lane >> 4) & 1) * 8;
    const unsigned* row = (const unsigned*)(X + m * stride);
    v8u a;
#pragma unroll
    for (int r = 0; r < 8; ++r) {
        int k = kbase + (r & 3) * 2 + (r >> 2) * 16 + half8;
        a[r] = row[k >> 1];
    }
    return __builtin_bit_cast(v16bf, a);
}

// B fragment from transposed weights WT[n][k], lane = N (mod 16):
//   k = kbase + (r&3)*2 + (r>>2)*16 + (lane>=16 ? 8 : 0)
__device__ __forceinline__ v16bf load_B(const unsigned short* WT, int stride,
                                        int n0, int kbase, int lane) {
    const int n = n0 + (lane & 15);
    const int half16 = ((lane >> 4) & 1) * 16;
    const unsigned* row = (const unsigned*)(WT + n * stride);
    v8u b;
#pragma unroll
    for (int r = 0; r < 8; ++r) {
        int k = kbase + (r & 3) * 2 + (r >> 2) * 16 + (half16 >> 1);
        b[r] = row[k >> 1];
    }
    return __builtin_bit_cast(v16bf, b);
}

template<int KCH, int NT>
__device__ __forceinline__ void gemm_tiles(const unsigned short* X, int xstride,
                                           const unsigned short* WT, int wstride,
                                           int lane, int m0, v8f* acc) {
    const int mrow = m0 + (lane & 15);
#pragma unroll
    for (int kc = 0; kc < KCH; ++kc) {
        v16bf a = load_A(X, xstride, mrow, kc * 32, lane);
#pragma unroll
        for (int t = 0; t < NT; ++t) {
            v16bf b = load_B(WT, wstride, t * 16, kc * 32, lane);
            acc[t] = wmma_bf16(a, b, acc[t]);
        }
    }
}

__device__ __forceinline__ void stage_bias(const float* __restrict__ B, float* S,
                                           int n, int tid) {
    for (int i = tid; i < n; i += 256) S[i] = B[i];
}

// bias + relu accumulators -> bf16 hidden in LDS, row-major [m][64]
__device__ __forceinline__ void store_h_relu(const v8f* acc, const float* b1s,
                                             unsigned short* hA, int lane, int m0) {
    const int half8 = ((lane >> 4) & 1) * 8;
#pragma unroll
    for (int t = 0; t < 4; ++t) {
        const int n = t * 16 + (lane & 15);
        const float bb = b1s[n];
#pragma unroll
        for (int r = 0; r < 8; ++r) {
            float x = fmaxf(acc[t][r] + bb, 0.f);
            hA[(m0 + r + half8) * 64 + n] = f32_bf16(x);
        }
    }
}

#define VZERO {0.f,0.f,0.f,0.f,0.f,0.f,0.f,0.f}

// ---------- kernels ----------

__global__ void zero_kernel(float4* __restrict__ p, int n4) {
    int i = blockIdx.x * blockDim.x + threadIdx.x;
    int s = gridDim.x * blockDim.x;
    float4 z = {0.f, 0.f, 0.f, 0.f};
    for (; i < n4; i += s) p[i] = z;
}

// one-time: W [K][N] f32 row-major -> transposed bf16 WT [nrows][stride], zero-padded
__global__ void prep_wT_kernel(const float* __restrict__ W,
                               unsigned short* __restrict__ WT,
                               int K, int N, int stride, int nrows) {
    int i = blockIdx.x * blockDim.x + threadIdx.x;
    int s = gridDim.x * blockDim.x;
    int tot = nrows * stride;
    for (; i < tot; i += s) {
        int n = i / stride, k = i % stride;
        float v = (k < K && n < N) ? W[k * N + n] : 0.f;
        WT[i] = f32_bf16(v);
    }
}

// MLP2: f32 [n,DIN] -> relu -> relu -> bf16 [n,64]   (DIN <= 32)
template<int DIN>
__global__ __launch_bounds__(256) void node_mlp2_kernel(
    const float* __restrict__ X,
    const unsigned short* __restrict__ W1T, const float* __restrict__ B1,
    const unsigned short* __restrict__ W2T, const float* __restrict__ B2,
    unsigned short* __restrict__ Yb, int nrows)
{
    __shared__ __align__(16) unsigned short xA[128 * 32];
    __shared__ __align__(16) unsigned short w1T[64 * 32];
    __shared__ __align__(16) unsigned short w2T[64 * 64];
    __shared__ __align__(16) unsigned short hA[128 * 64];
    __shared__ float b1s[64], b2s[64];
    const int tid = threadIdx.x;
    const int row0 = blockIdx.x * 128;

    async_stage(W1T, w1T, 64 * 32, tid);
    async_stage(W2T, w2T, 64 * 64, tid);
    for (int i = tid; i < 128 * 32; i += 256) {
        int m = i >> 5, k = i & 31;
        int g = row0 + m;
        float v = (g < nrows && k < DIN) ? X[(size_t)g * DIN + k] : 0.f;
        xA[i] = f32_bf16(v);
    }
    stage_bias(B1, b1s, 64, tid);
    stage_bias(B2, b2s, 64, tid);
    wait_async0();
    __syncthreads();

    const int lane = tid & 31, m0 = (tid >> 5) * 16;
    v8f z = VZERO;
    v8f acc[4] = {z, z, z, z};
    gemm_tiles<1, 4>(xA, 32, w1T, 32, lane, m0, acc);
    store_h_relu(acc, b1s, hA, lane, m0);
    __syncthreads();
    v8f acc2[4] = {z, z, z, z};
    gemm_tiles<2, 4>(hA, 64, w2T, 64, lane, m0, acc2);

    const int half8 = ((lane >> 4) & 1) * 8;
#pragma unroll
    for (int r = 0; r < 8; ++r) {
        int g = row0 + m0 + r + half8;
        if (g < nrows) {
#pragma unroll
            for (int t = 0; t < 4; ++t) {
                int n = t * 16 + (lane & 15);
                Yb[(size_t)g * 64 + n] = f32_bf16(fmaxf(acc2[t][r] + b2s[n], 0.f));
            }
        }
    }
}

// Fused edge MLP + segment-sum scatter, bf16 feature gathers via async-to-LDS:
//   x = [U[uidx], V[vidx], eval]  (K=129, padded to 160)
//   agg[uidx] += relu(relu(x@W1+b1)@W2+b2)   via global f32 atomics
__global__ __launch_bounds__(256) void edge_mlp_scatter_kernel(
    const unsigned short* __restrict__ U, const unsigned short* __restrict__ V,
    const int* __restrict__ uidx, const int* __restrict__ vidx,
    const float* __restrict__ eval,
    const unsigned short* __restrict__ W1T, const float* __restrict__ B1,
    const unsigned short* __restrict__ W2T, const float* __restrict__ B2,
    float* __restrict__ agg, int nE)
{
    __shared__ __align__(16) unsigned short xA[128 * 160];
    __shared__ __align__(16) unsigned short w1T[64 * 160];
    __shared__ __align__(16) unsigned short w2T[64 * 64];
    __shared__ __align__(16) unsigned short hA[128 * 64];
    __shared__ float b1s[64], b2s[64];
    __shared__ int su[128], sv[128];
    const int tid = threadIdx.x;
    const int e0 = blockIdx.x * 128;

    async_stage(W1T, w1T, 64 * 160, tid);
    async_stage(W2T, w2T, 64 * 64, tid);
    if (tid < 128) {
        int e = e0 + tid;
        bool ok = e < nE;
        su[tid] = ok ? uidx[e] : -1;
        sv[tid] = ok ? vidx[e] : 0;
        float ev = ok ? eval[e] : 0.f;
        unsigned short* row = xA + tid * 160;
        row[128] = f32_bf16(ev);
        for (int k = 129; k < 160; ++k) row[k] = 0;
    }
    stage_bias(B1, b1s, 64, tid);
    stage_bias(B2, b2s, 64, tid);
    __syncthreads();                       // su/sv visible

    const unsigned xab = (unsigned)(uintptr_t)xA;
    for (int i = tid; i < 128 * 16; i += 256) {
        int e = i >> 4, cq = i & 15;
        int part = cq >> 3, q = cq & 7;    // part 0 = U side, 1 = V side
        int ui = su[e]; if (ui < 0) ui = 0;
        int vi = sv[e];
        const unsigned short* src = (part == 0) ? (U + (size_t)ui * 64 + q * 8)
                                                : (V + (size_t)vi * 64 + q * 8);
        async_copy16(xab + 2u * (unsigned)(e * 160 + part * 64 + q * 8), src);
    }
    wait_async0();
    __syncthreads();

    const int lane = tid & 31, m0 = (tid >> 5) * 16;
    v8f z = VZERO;
    v8f acc[4] = {z, z, z, z};
    gemm_tiles<5, 4>(xA, 160, w1T, 160, lane, m0, acc);      // 20 wmma
    store_h_relu(acc, b1s, hA, lane, m0);
    __syncthreads();
    v8f acc2[4] = {z, z, z, z};
    gemm_tiles<2, 4>(hA, 64, w2T, 64, lane, m0, acc2);       // 8 wmma

    const int half8 = ((lane >> 4) & 1) * 8;
#pragma unroll
    for (int r = 0; r < 8; ++r) {
        int m = m0 + r + half8;
        int ui = su[m];
        if (ui >= 0) {
#pragma unroll
            for (int t = 0; t < 4; ++t) {
                int n = t * 16 + (lane & 15);
                float x = fmaxf(acc2[t][r] + b2s[n], 0.f);
                unsafeAtomicAdd(&agg[(size_t)ui * 64 + n], x);   // global_atomic_add_f32
            }
        }
    }
}

// MLP2 over concat([U(bf16), A(f32 agg)]): Din = 128 -> bf16 out
__global__ __launch_bounds__(256) void concat_mlp2_kernel(
    const unsigned short* __restrict__ U, const float* __restrict__ A,
    const unsigned short* __restrict__ W1T, const float* __restrict__ B1,
    const unsigned short* __restrict__ W2T, const float* __restrict__ B2,
    unsigned short* __restrict__ Yb, int nrows)
{
    __shared__ __align__(16) unsigned short xA[128 * 128];
    __shared__ __align__(16) unsigned short w1T[64 * 128];
    __shared__ __align__(16) unsigned short w2T[64 * 64];
    __shared__ __align__(16) unsigned short hA[128 * 64];
    __shared__ float b1s[64], b2s[64];
    const int tid = threadIdx.x;
    const int row0 = blockIdx.x * 128;

    async_stage(W1T, w1T, 64 * 128, tid);
    async_stage(W2T, w2T, 64 * 64, tid);
    const unsigned xab = (unsigned)(uintptr_t)xA;
    for (int i = tid; i < 128 * 8; i += 256) {           // U half, async 16B
        int m = i >> 3, q = i & 7;
        int g = row0 + m; if (g >= nrows) g = nrows - 1;
        async_copy16(xab + 2u * (unsigned)(m * 128 + q * 8),
                     U + (size_t)g * 64 + q * 8);
    }
    for (int i = tid; i < 128 * 16; i += 256) {          // agg half, f32->bf16 packed
        int m = i >> 4, q = i & 15;
        int g = row0 + m;
        float4 a4 = {0.f, 0.f, 0.f, 0.f};
        if (g < nrows) a4 = *(const float4*)&A[(size_t)g * 64 + q * 4];
        uint2 p; p.x = pack_bf16(a4.x, a4.y); p.y = pack_bf16(a4.z, a4.w);
        *(uint2*)&xA[m * 128 + 64 + q * 4] = p;
    }
    stage_bias(B1, b1s, 64, tid);
    stage_bias(B2, b2s, 64, tid);
    wait_async0();
    __syncthreads();

    const int lane = tid & 31, m0 = (tid >> 5) * 16;
    v8f z = VZERO;
    v8f acc[4] = {z, z, z, z};
    gemm_tiles<4, 4>(xA, 128, w1T, 128, lane, m0, acc);      // 16 wmma
    store_h_relu(acc, b1s, hA, lane, m0);
    __syncthreads();
    v8f acc2[4] = {z, z, z, z};
    gemm_tiles<2, 4>(hA, 64, w2T, 64, lane, m0, acc2);

    const int half8 = ((lane >> 4) & 1) * 8;
#pragma unroll
    for (int r = 0; r < 8; ++r) {
        int g = row0 + m0 + r + half8;
        if (g < nrows) {
#pragma unroll
            for (int t = 0; t < 4; ++t) {
                int n = t * 16 + (lane & 15);
                Yb[(size_t)g * 64 + n] = f32_bf16(fmaxf(acc2[t][r] + b2s[n], 0.f));
            }
        }
    }
}

// tail: sigmoid(relu(X@W1+b1)@W2+b2), K=8 outputs (N padded to 16), f32 out
__global__ __launch_bounds__(256) void tail_kernel(
    const unsigned short* __restrict__ Xb,
    const unsigned short* __restrict__ W1T, const float* __restrict__ B1,
    const unsigned short* __restrict__ W2T, const float* __restrict__ B2,
    float* __restrict__ OUT, int nrows)
{
    __shared__ __align__(16) unsigned short xA[128 * 64];
    __shared__ __align__(16) unsigned short w1T[64 * 64];
    __shared__ __align__(16) unsigned short w2T[16 * 64];
    __shared__ __align__(16) unsigned short hA[128 * 64];
    __shared__ float b1s[64], b2s[16];
    const int tid = threadIdx.x;
    const int row0 = blockIdx.x * 128;

    async_stage(W1T, w1T, 64 * 64, tid);
    async_stage(W2T, w2T, 16 * 64, tid);
    const unsigned xab = (unsigned)(uintptr_t)xA;
    for (int i = tid; i < 128 * 8; i += 256) {
        int m = i >> 3, q = i & 7;
        int g = row0 + m; if (g >= nrows) g = nrows - 1;
        async_copy16(xab + 2u * (unsigned)(m * 64 + q * 8),
                     Xb + (size_t)g * 64 + q * 8);
    }
    stage_bias(B1, b1s, 64, tid);
    for (int i = tid; i < 16; i += 256) b2s[i] = (i < 8) ? B2[i] : 0.f;
    wait_async0();
    __syncthreads();

    const int lane = tid & 31, m0 = (tid >> 5) * 16;
    v8f z = VZERO;
    v8f acc[4] = {z, z, z, z};
    gemm_tiles<2, 4>(xA, 64, w1T, 64, lane, m0, acc);
    store_h_relu(acc, b1s, hA, lane, m0);
    __syncthreads();
    v8f acc2[1] = {z};
    gemm_tiles<2, 1>(hA, 64, w2T, 64, lane, m0, acc2);

    const int half8 = ((lane >> 4) & 1) * 8;
    const int n = lane & 15;
    if (n < 8) {
#pragma unroll
        for (int r = 0; r < 8; ++r) {
            int g = row0 + m0 + r + half8;
            if (g < nrows) {
                float x = acc2[0][r] + b2s[n];
                OUT[(size_t)g * 8 + n] = 1.f / (1.f + __expf(-x));
            }
        }
    }
}

// ---------- launch ----------

extern "C" void kernel_launch(void* const* d_in, const int* in_sizes, int n_in,
                              void* d_out, int out_size, void* d_ws, size_t ws_size,
                              hipStream_t stream) {
    const float* v        = (const float*)d_in[0];
    const float* c        = (const float*)d_in[1];
    const int*   cons_idx = (const int*)d_in[2];
    const int*   var_idx  = (const int*)d_in[3];
    const float* e_val    = (const float*)d_in[4];
    auto P = [&](int i) { return (const float*)d_in[i]; };

    const int NV = in_sizes[0] / 19;
    const int NC = in_sizes[1] / 5;
    const int NE = in_sizes[4];

    char* w = (char*)d_ws;
    auto alloc = [&](size_t bytes) { void* p = (void*)w; w += (bytes + 255) & ~(size_t)255; return p; };
    float* aggC = (float*)alloc((size_t)NC * 64 * 4);
    float* aggV = (float*)alloc((size_t)NV * 64 * 4);
    unsigned short* v1b = (unsigned short*)alloc((size_t)NV * 64 * 2);
    unsigned short* c1b = (unsigned short*)alloc((size_t)NC * 64 * 2);
    unsigned short* c2b = (unsigned short*)alloc((size_t)NC * 64 * 2);
    unsigned short* v2b = (unsigned short*)alloc((size_t)NV * 64 * 2);
    unsigned short* evw1T = (unsigned short*)alloc(64 * 32 * 2);
    unsigned short* evw2T = (unsigned short*)alloc(64 * 64 * 2);
    unsigned short* ecw1T = (unsigned short*)alloc(64 * 32 * 2);
    unsigned short* ecw2T = (unsigned short*)alloc(64 * 64 * 2);
    unsigned short* cgw1T = (unsigned short*)alloc(64 * 160 * 2);
    unsigned short* cgw2T = (unsigned short*)alloc(64 * 64 * 2);
    unsigned short* cfw1T = (unsigned short*)alloc(64 * 128 * 2);
    unsigned short* cfw2T = (unsigned short*)alloc(64 * 64 * 2);
    unsigned short* vgw1T = (unsigned short*)alloc(64 * 160 * 2);
    unsigned short* vgw2T = (unsigned short*)alloc(64 * 64 * 2);
    unsigned short* vfw1T = (unsigned short*)alloc(64 * 128 * 2);
    unsigned short* vfw2T = (unsigned short*)alloc(64 * 64 * 2);
    unsigned short* tw1T  = (unsigned short*)alloc(64 * 64 * 2);
    unsigned short* tw2T  = (unsigned short*)alloc(16 * 64 * 2);

    const dim3 blk(256);
    // one-time weight transpose + bf16 conversion
    prep_wT_kernel<<<8, blk, 0, stream>>>(P(5),  evw1T, 19, 64, 32, 64);
    prep_wT_kernel<<<8, blk, 0, stream>>>(P(7),  evw2T, 64, 64, 64, 64);
    prep_wT_kernel<<<8, blk, 0, stream>>>(P(9),  ecw1T, 5, 64, 32, 64);
    prep_wT_kernel<<<8, blk, 0, stream>>>(P(11), ecw2T, 64, 64, 64, 64);
    prep_wT_kernel<<<8, blk, 0, stream>>>(P(13), cgw1T, 129, 64, 160, 64);
    prep_wT_kernel<<<8, blk, 0, stream>>>(P(15), cgw2T, 64, 64, 64, 64);
    prep_wT_kernel<<<8, blk, 0, stream>>>(P(17), cfw1T, 128, 64, 128, 64);
    prep_wT_kernel<<<8, blk, 0, stream>>>(P(19), cfw2T, 64, 64, 64, 64);
    prep_wT_kernel<<<8, blk, 0, stream>>>(P(21), vgw1T, 129, 64, 160, 64);
    prep_wT_kernel<<<8, blk, 0, stream>>>(P(23), vgw2T, 64, 64, 64, 64);
    prep_wT_kernel<<<8, blk, 0, stream>>>(P(25), vfw1T, 128, 64, 128, 64);
    prep_wT_kernel<<<8, blk, 0, stream>>>(P(27), vfw2T, 64, 64, 64, 64);
    prep_wT_kernel<<<8, blk, 0, stream>>>(P(29), tw1T, 64, 64, 64, 64);
    prep_wT_kernel<<<8, blk, 0, stream>>>(P(31), tw2T, 64, 8, 64, 16);

    // node embeddings (bf16 out)
    node_mlp2_kernel<19><<<(NV + 127) / 128, blk, 0, stream>>>(v, evw1T, P(6), evw2T, P(8), v1b, NV);
    node_mlp2_kernel<5> <<<(NC + 127) / 128, blk, 0, stream>>>(c, ecw1T, P(10), ecw2T, P(12), c1b, NC);
    // c-side half-conv
    zero_kernel<<<512, blk, 0, stream>>>((float4*)aggC, NC * 16);
    edge_mlp_scatter_kernel<<<(NE + 127) / 128, blk, 0, stream>>>(
        c1b, v1b, cons_idx, var_idx, e_val, cgw1T, P(14), cgw2T, P(16), aggC, NE);
    concat_mlp2_kernel<<<(NC + 127) / 128, blk, 0, stream>>>(
        c1b, aggC, cfw1T, P(18), cfw2T, P(20), c2b, NC);
    // v-side half-conv
    zero_kernel<<<512, blk, 0, stream>>>((float4*)aggV, NV * 16);
    edge_mlp_scatter_kernel<<<(NE + 127) / 128, blk, 0, stream>>>(
        v1b, c2b, var_idx, cons_idx, e_val, vgw1T, P(22), vgw2T, P(24), aggV, NE);
    concat_mlp2_kernel<<<(NV + 127) / 128, blk, 0, stream>>>(
        v1b, aggV, vfw1T, P(26), vfw2T, P(28), v2b, NV);
    // tail
    tail_kernel<<<(NV + 127) / 128, blk, 0, stream>>>(
        v2b, tw1T, P(30), tw2T, P(32), (float*)d_out, NV);
}
